// ScaledDotProductAttention_67044439491017
// MI455X (gfx1250) — matrix operand order repaired and checked by reading, not compiled, not run on hardware
//
#include <hip/hip_runtime.h>

// Flash-attention forward, causal, fp32 I/O, bf16 WMMA compute (f32 accum).
// B=4 H=16 S=2048 D=64. Grid: (S/64, B*H). Block: 128 threads = 4 wave32.
// - double-buffered K/V LDS tiles, 1 barrier/iter
// - unmasked main loop + masked tail (template<bool MASK>); fully-masked
//   tail tiles skipped per-wave (wave-uniform branch)
// - wave-level running max (one 32-lane butterfly per tile); rescale is
//   UNconditional (corr==1.0 when max unchanged) so acc stays a single SSA
//   chain and PV WMMAs accumulate fully in place (no register copies)
// - row-sum (softmax denominator) accumulated by an extra WMMA vs all-ones B
// - exp2-domain softmax (log2e folded into Q scale), v_perm bf16 packing

namespace {
constexpr int S  = 2048;
constexpr int D  = 64;
constexpr int BQ = 64;   // queries per workgroup (4 waves x 16 rows)
constexpr int BK = 32;   // keys per iteration
}

typedef __attribute__((ext_vector_type(16))) __bf16 bf16x16;
typedef __attribute__((ext_vector_type(8)))  float  f32x8;

union Frag { uint4 u[2]; bf16x16 v; };

// pack two f32 -> two bf16 (round-to-nearest, ties-away) in one v_perm_b32
__device__ __forceinline__ unsigned int pkbf(float a, float b) {
  unsigned ua = __float_as_uint(a) + 0x8000u;
  unsigned ub = __float_as_uint(b) + 0x8000u;
  return __builtin_amdgcn_perm(ub, ua, 0x07060302u);   // {hi16(b), hi16(a)}
}
__device__ __forceinline__ unsigned short bf1(float a) {
  return (unsigned short)((__float_as_uint(a) + 0x8000u) >> 16);
}

__device__ __forceinline__ f32x8 wmma_bf16(const Frag& a, const Frag& b, f32x8 c) {
  return __builtin_amdgcn_wmma_f32_16x16x32_bf16(false, a.v, false, b.v,
                                                 (short)0, c, false, false);
}

// One 32-key tile: scores (2 WMMA-chains), online softmax, O += P@V (4 WMMA),
// l += P@ones (1 WMMA).  acc[0..3]=O d-tiles, acc[4]=running row-sum l.
template<bool MASK>
__device__ __forceinline__ void compute_tile(
    int kb, int qw, int half, int l16,
    const unsigned short (*sKb)[D], const unsigned short (*sVtb)[BK],
    unsigned short (*sPw)[BK],
    const Frag (&qa)[2], f32x8 (&acc)[5], float& m_run)
{
  const f32x8 vzero = {0.f,0.f,0.f,0.f,0.f,0.f,0.f,0.f};
  f32x8 s[2];
  #pragma unroll
  for (int t = 0; t < 2; ++t) {
    Frag b0, b1;
    b0.u[0] = *(const uint4*)&sKb[t*16 + l16][half*16];
    b0.u[1] = *(const uint4*)&sKb[t*16 + l16][half*16 + 8];
    b1.u[0] = *(const uint4*)&sKb[t*16 + l16][32 + half*16];
    b1.u[1] = *(const uint4*)&sKb[t*16 + l16][32 + half*16 + 8];
    s[t] = wmma_bf16(qa[0], b0, vzero);
    s[t] = wmma_bf16(qa[1], b1, s[t]);
  }

  float a0[8], a1[8];
  #pragma unroll
  for (int r = 0; r < 8; ++r) {
    a0[r] = s[0][r];
    a1[r] = s[1][r];
    if (MASK) {
      const int row = qw + half*8 + r;
      a0[r] = (kb      + l16 > row) ? -1.0e30f : a0[r];
      a1[r] = (kb + 16 + l16 > row) ? -1.0e30f : a1[r];
    }
  }

  // wave-wide max bound: per-lane max of 16 elems + one 32-lane butterfly
  float mx = fmaxf(a0[0], a1[0]);
  #pragma unroll
  for (int r = 1; r < 8; ++r) mx = fmaxf(mx, fmaxf(a0[r], a1[r]));
  #pragma unroll
  for (int off = 1; off < 32; off <<= 1)
    mx = fmaxf(mx, __shfl_xor(mx, off, 32));

  // unconditional, wave-uniform rescale: corr == 1.0 when max unchanged.
  // Keeping this branch-free keeps acc a single SSA chain -> in-place WMMA.
  const float mn   = fmaxf(m_run, mx);
  const float corr = __builtin_amdgcn_exp2f(m_run - mn);
  m_run = mn;
  #pragma unroll
  for (int t = 0; t < 5; ++t) {
    #pragma unroll
    for (int r = 0; r < 8; ++r) acc[t][r] *= corr;
  }

  #pragma unroll
  for (int r = 0; r < 8; ++r) {
    const int prow = half*8 + r;
    sPw[prow][l16]      = bf1(__builtin_amdgcn_exp2f(a0[r] - mn));
    sPw[prow][l16 + 16] = bf1(__builtin_amdgcn_exp2f(a1[r] - mn));
  }

  Frag pa;                                // same-wave DS ops are in order
  pa.u[0] = *(const uint4*)&sPw[l16][half*16];
  pa.u[1] = *(const uint4*)&sPw[l16][half*16 + 8];
  #pragma unroll
  for (int t = 0; t < 4; ++t) {
    Frag vb;
    vb.u[0] = *(const uint4*)&sVtb[t*16 + l16][half*16];
    vb.u[1] = *(const uint4*)&sVtb[t*16 + l16][half*16 + 8];
    acc[t] = wmma_bf16(pa, vb, acc[t]);
  }
  Frag ones;                              // bf16 1.0 everywhere
  ones.u[0] = make_uint4(0x3F803F80u, 0x3F803F80u, 0x3F803F80u, 0x3F803F80u);
  ones.u[1] = ones.u[0];
  acc[4] = wmma_bf16(pa, ones, acc[4]);   // row sums, broadcast to all lanes
}

__global__ __launch_bounds__(128)
void fa_fwd_causal(const float* __restrict__ Q, const float* __restrict__ K,
                   const float* __restrict__ V, float* __restrict__ O)
{
  __shared__ alignas(16) unsigned short sK [2][BK][D];   // bf16 K tiles [key][d]
  __shared__ alignas(16) unsigned short sVt[2][D][BK];   // bf16 V^T tiles [d][key]
  __shared__ alignas(16) unsigned short sP [4][16][BK];  // per-wave P scratch

  const int tid  = threadIdx.x;
  const int lane = tid & 31;
  const int wave = tid >> 5;
  const int half = lane >> 4;
  const int l16  = lane & 15;

  const int    q0   = blockIdx.x * BQ;
  const size_t base = (size_t)blockIdx.y * S * D;
  const int    qw   = q0 + wave * 16;

  // ---- Q A-fragments, scaled by log2(e)/sqrt(D) (exp2-domain softmax) ----
  Frag qa[2];
  {
    const float  sc = 0.125f * 1.4426950408889634f;
    const float* qp = Q + base + (size_t)(qw + l16) * D;
    #pragma unroll
    for (int c = 0; c < 2; ++c) {
      const int b1 = c * 32 + half * 16;
      const int b2 = b1 + 8;
      float4 f0 = *(const float4*)(qp + b1);
      float4 f1 = *(const float4*)(qp + b1 + 4);
      float4 f2 = *(const float4*)(qp + b2);
      float4 f3 = *(const float4*)(qp + b2 + 4);
      qa[c].u[0] = make_uint4(pkbf(f0.x*sc, f0.y*sc), pkbf(f0.z*sc, f0.w*sc),
                              pkbf(f1.x*sc, f1.y*sc), pkbf(f1.z*sc, f1.w*sc));
      qa[c].u[1] = make_uint4(pkbf(f2.x*sc, f2.y*sc), pkbf(f2.z*sc, f2.w*sc),
                              pkbf(f3.x*sc, f3.y*sc), pkbf(f3.z*sc, f3.w*sc));
    }
  }

  const f32x8 vzero = {0.f,0.f,0.f,0.f,0.f,0.f,0.f,0.f};
  f32x8 acc[5];
  #pragma unroll
  for (int t = 0; t < 5; ++t) acc[t] = vzero;
  float m_run = -1.0e30f;

  // cooperative staging: thread owns 2 adjacent keys x 8 dims
  const int key0  = (tid & 15) * 2;     // 0,2,..,30
  const int dseg8 = (tid >> 4) * 8;     // 0,8,..,56

  float4 kr[4], vr[4];                  // staging registers (double-buffer feed)
  auto load_tile = [&](int it) {
    const float* kp = K + base + (size_t)(it * BK + key0) * D + dseg8;
    const float* vp = V + base + (size_t)(it * BK + key0) * D + dseg8;
    kr[0] = *(const float4*)(kp);      kr[1] = *(const float4*)(kp + 4);
    kr[2] = *(const float4*)(kp + D);  kr[3] = *(const float4*)(kp + D + 4);
    vr[0] = *(const float4*)(vp);      vr[1] = *(const float4*)(vp + 4);
    vr[2] = *(const float4*)(vp + D);  vr[3] = *(const float4*)(vp + D + 4);
  };
  auto store_tile = [&](int pb) {
    *(uint4*)&sK[pb][key0][dseg8] =
        make_uint4(pkbf(kr[0].x, kr[0].y), pkbf(kr[0].z, kr[0].w),
                   pkbf(kr[1].x, kr[1].y), pkbf(kr[1].z, kr[1].w));
    *(uint4*)&sK[pb][key0 + 1][dseg8] =
        make_uint4(pkbf(kr[2].x, kr[2].y), pkbf(kr[2].z, kr[2].w),
                   pkbf(kr[3].x, kr[3].y), pkbf(kr[3].z, kr[3].w));
    const float va[8] = {vr[0].x, vr[0].y, vr[0].z, vr[0].w,
                         vr[1].x, vr[1].y, vr[1].z, vr[1].w};
    const float vb[8] = {vr[2].x, vr[2].y, vr[2].z, vr[2].w,
                         vr[3].x, vr[3].y, vr[3].z, vr[3].w};
    #pragma unroll
    for (int j = 0; j < 8; ++j)        // transposed store, 2 keys packed -> b32
      *(unsigned int*)&sVt[pb][dseg8 + j][key0] = pkbf(va[j], vb[j]);
  };

  const int nit = (q0 + BQ) / BK;       // causal: keys < q0+64
  load_tile(0);
  store_tile(0);
  __syncthreads();

  for (int it = 0; it < nit; ++it) {
    const int  kb        = it * BK;
    const bool have_next = (it + 1) < nit;
    if (have_next) load_tile(it + 1);   // global loads fly during compute
    if (it + 2 < nit) {                 // uniform branch: L2 prefetch 2 ahead
      __builtin_prefetch(K + base + (size_t)((it + 2) * BK + key0) * D + dseg8, 0, 0);
      __builtin_prefetch(V + base + (size_t)((it + 2) * BK + key0) * D + dseg8, 0, 0);
    }
    if (kb < q0)                        // block-uniform: no mask VALU in main loop
      compute_tile<false>(kb, qw, half, l16, sK[it & 1], sVt[it & 1], sP[wave],
                          qa, acc, m_run);
    else if (kb <= qw + 15)             // wave-uniform: skip fully-masked tiles
      compute_tile<true>(kb, qw, half, l16, sK[it & 1], sVt[it & 1], sP[wave],
                         qa, acc, m_run);
    if (have_next) store_tile((it + 1) & 1);
    __syncthreads();
  }

  // ---- epilogue: O = acc / l  (l broadcast in every lane via ones-WMMA) ----
  #pragma unroll
  for (int r = 0; r < 8; ++r) {
    const float inv = 1.0f / acc[4][r];
    float* op = O + base + (size_t)(qw + half*8 + r) * D + l16;
    op[0]  = acc[0][r] * inv;
    op[16] = acc[1][r] * inv;
    op[32] = acc[2][r] * inv;
    op[48] = acc[3][r] * inv;
  }
}

extern "C" void kernel_launch(void* const* d_in, const int* in_sizes, int n_in,
                              void* d_out, int out_size, void* d_ws, size_t ws_size,
                              hipStream_t stream) {
  const float* Q = (const float*)d_in[0];
  const float* K = (const float*)d_in[1];
  const float* V = (const float*)d_in[2];
  // d_in[3] (mask) unused: causality computed analytically in-kernel.
  float* O = (float*)d_out;
  const int BH = in_sizes[0] / (S * D);   // B*H = 64
  dim3 grid(S / BQ, BH);
  fa_fwd_causal<<<grid, dim3(128), 0, stream>>>(Q, K, V, O);
}
